// NeuralNetwork_10153302688109
// MI455X (gfx1250) — compile-verified, matrix operand order checked by reading, not run on hardware
//
#include <hip/hip_runtime.h>

// ---------------------------------------------------------------------------
// MLP 4 -> 128 -> 128 -> 128 -> 2 (tanh between layers), fp32 in/out,
// f16 WMMA (V_WMMA_F32_16X16X32_F16) with fp32 accumulation on gfx1250.
// ---------------------------------------------------------------------------

typedef __attribute__((ext_vector_type(16))) _Float16 v16h;
typedef __attribute__((ext_vector_type(8)))  _Float16 v8h;
typedef __attribute__((ext_vector_type(8)))  float    v8f;

#define HID 128

__device__ __forceinline__ v8f wmma16(v16h a, v16h b, v8f c) {
    // (neg_a, A, neg_b, B, c_mod, C, reuse_a, reuse_b)
    return __builtin_amdgcn_wmma_f32_16x16x32_f16(false, a, false, b,
                                                  (short)0, c, false, false);
}

// tanh(x) = 1 - 2/(exp(2x)+1): v_exp_f32 + v_rcp_f32, saturates correctly.
__device__ __forceinline__ float fast_tanh(float x) {
    float e = __expf(2.0f * x);
    return 1.0f - 2.0f * __builtin_amdgcn_rcpf(e + 1.0f);
}

// Load one 16x32-K f16 operand chunk for this lane from a row-major
// [row][rowstride] f16 LDS matrix, following the CDNA5 16-bit A layout:
//   lanes 0-15 : K = kc*32 + {0..7} (v0-3) and kc*32 + {16..23} (v4-7)
//   lanes 16-31: K = kc*32 + {8..15}       and kc*32 + {24..31}
__device__ __forceinline__ v16h load_mat(const _Float16* __restrict__ base,
                                         int row, int rowstride,
                                         int kc, bool hiHalf) {
    const _Float16* p = base + row * rowstride + kc * 32 + (hiHalf ? 8 : 0);
    v8h lo = *(const v8h*)(p);        // ds_load_b128
    v8h hi = *(const v8h*)(p + 16);   // ds_load_b128
    return __builtin_shufflevector(lo, hi, 0, 1, 2, 3, 4, 5, 6, 7,
                                   8, 9, 10, 11, 12, 13, 14, 15);
}

__global__ __launch_bounds__(256)
void mlp_wmma_kernel(const float* __restrict__ x,  const float* __restrict__ y,
                     const float* __restrict__ c1, const float* __restrict__ c2,
                     const float* __restrict__ W1, const float* __restrict__ b1,
                     const float* __restrict__ W2, const float* __restrict__ b2,
                     const float* __restrict__ W3, const float* __restrict__ b3,
                     const float* __restrict__ W4, const float* __restrict__ b4,
                     float* __restrict__ out, int npts) {
    // Weights transposed to [out][in] f16 so B-operand loads match load_mat().
    __shared__ _Float16 sW1[HID * 32];        // [n][k], k 4 -> padded 32
    __shared__ _Float16 sW2[HID * HID];       // [n][k]
    __shared__ _Float16 sW3[HID * HID];       // [n][k]
    __shared__ _Float16 sW4[16 * HID];        // [n][k], n 2 -> padded 16
    __shared__ float    sB1[HID], sB2[HID], sB3[HID], sB4[16];
    __shared__ _Float16 sHa[8][16 * HID];     // per-wave activation ping
    __shared__ _Float16 sHb[8][16 * HID];     // per-wave activation pong

    const int t = threadIdx.x;

    // ---- one-time weight staging (f32 -> f16, transposed) ----
    for (int i = t; i < HID * 32; i += 256) {
        int n = i >> 5, k = i & 31;
        sW1[i] = (k < 4) ? (_Float16)W1[k * HID + n] : (_Float16)0.0f;
    }
    for (int i = t; i < HID * HID; i += 256) {
        int k = i >> 7, n = i & 127;               // coalesced global read
        sW2[n * HID + k] = (_Float16)W2[i];
        sW3[n * HID + k] = (_Float16)W3[i];
    }
    for (int i = t; i < 16 * HID; i += 256) {
        int n = i >> 7, k = i & 127;
        sW4[i] = (n < 2) ? (_Float16)W4[k * 2 + n] : (_Float16)0.0f;
    }
    if (t < HID) { sB1[t] = b1[t]; sB2[t] = b2[t]; sB3[t] = b3[t]; }
    if (t < 16)  { sB4[t] = (t < 2) ? b4[t] : 0.0f; }
    __syncthreads();

    const int lane  = t & 31;
    const int wave  = t >> 5;
    const int lo16  = lane & 15;
    const bool hiH  = lane >= 16;
    _Float16* Ha = &sHa[wave][0];
    _Float16* Hb = &sHb[wave][0];

    const int ntiles = (npts + 15) >> 4;
    const int nwaves = gridDim.x * 8;

    for (int tile = blockIdx.x * 8 + wave; tile < ntiles; tile += nwaves) {
        const int p0 = tile << 4;

        // ------- Layer 1: A = inputs 16x4, zero-padded to 16x32 -------
        v16h a1 = (v16h)(_Float16)0.0f;
        {
            int pidx = p0 + lo16;
            if (!hiH && pidx < npts) {            // K=0..3 live in lanes 0-15
                a1[0] = (_Float16)x[pidx];
                a1[1] = (_Float16)y[pidx];
                a1[2] = (_Float16)c1[pidx];
                a1[3] = (_Float16)c2[pidx];
            }
        }
        const int mbase = hiH ? 8 : 0;
        for (int nt = 0; nt < 8; ++nt) {
            v8f acc = {};
            v16h b = load_mat(sW1, nt * 16 + lo16, 32, 0, hiH);
            acc = wmma16(a1, b, acc);
            float bias = sB1[nt * 16 + lo16];
            int n = nt * 16 + lo16;
            #pragma unroll
            for (int i = 0; i < 8; ++i)
                Ha[(mbase + i) * HID + n] = (_Float16)fast_tanh(acc[i] + bias);
        }

        // ------- Layer 2: Ha (16x128) @ W2 -> Hb -------
        {
            v16h A[4];
            #pragma unroll
            for (int kc = 0; kc < 4; ++kc) A[kc] = load_mat(Ha, lo16, HID, kc, hiH);
            for (int nt = 0; nt < 8; ++nt) {
                v8f acc = {};
                #pragma unroll
                for (int kc = 0; kc < 4; ++kc) {
                    v16h b = load_mat(sW2, nt * 16 + lo16, HID, kc, hiH);
                    acc = wmma16(A[kc], b, acc);
                }
                float bias = sB2[nt * 16 + lo16];
                int n = nt * 16 + lo16;
                #pragma unroll
                for (int i = 0; i < 8; ++i)
                    Hb[(mbase + i) * HID + n] = (_Float16)fast_tanh(acc[i] + bias);
            }
        }

        // ------- Layer 3: Hb @ W3 -> Ha -------
        {
            v16h A[4];
            #pragma unroll
            for (int kc = 0; kc < 4; ++kc) A[kc] = load_mat(Hb, lo16, HID, kc, hiH);
            for (int nt = 0; nt < 8; ++nt) {
                v8f acc = {};
                #pragma unroll
                for (int kc = 0; kc < 4; ++kc) {
                    v16h b = load_mat(sW3, nt * 16 + lo16, HID, kc, hiH);
                    acc = wmma16(A[kc], b, acc);
                }
                float bias = sB3[nt * 16 + lo16];
                int n = nt * 16 + lo16;
                #pragma unroll
                for (int i = 0; i < 8; ++i)
                    Ha[(mbase + i) * HID + n] = (_Float16)fast_tanh(acc[i] + bias);
            }
        }

        // ------- Layer 4: Ha @ W4 (padded 128x16, n=0,1 valid) -------
        {
            v8f acc = {};
            #pragma unroll
            for (int kc = 0; kc < 4; ++kc) {
                v16h a = load_mat(Ha, lo16, HID, kc, hiH);
                v16h b = load_mat(sW4, lo16, HID, kc, hiH);
                acc = wmma16(a, b, acc);
            }
            if (lo16 < 2) {
                float bias = sB4[lo16];
                #pragma unroll
                for (int i = 0; i < 8; ++i) {
                    int p = p0 + mbase + i;
                    if (p < npts) out[p * 2 + lo16] = acc[i] + bias;
                }
            }
        }
    }
}

extern "C" void kernel_launch(void* const* d_in, const int* in_sizes, int n_in,
                              void* d_out, int out_size, void* d_ws, size_t ws_size,
                              hipStream_t stream) {
    const float* x  = (const float*)d_in[0];
    const float* y  = (const float*)d_in[1];
    const float* c1 = (const float*)d_in[2];
    const float* c2 = (const float*)d_in[3];
    const float* W1 = (const float*)d_in[4];
    const float* b1 = (const float*)d_in[5];
    const float* W2 = (const float*)d_in[6];
    const float* b2 = (const float*)d_in[7];
    const float* W3 = (const float*)d_in[8];
    const float* b3 = (const float*)d_in[9];
    const float* W4 = (const float*)d_in[10];
    const float* b4 = (const float*)d_in[11];
    float* out = (float*)d_out;

    int npts  = in_sizes[0];                 // 524288
    int tiles = (npts + 15) >> 4;            // 16 points per wave-tile
    int blocks = (tiles + 7) / 8;            // 8 waves per block
    if (blocks > 1024) blocks = 1024;        // grid-stride, amortize staging
    if (blocks < 1) blocks = 1;

    mlp_wmma_kernel<<<blocks, 256, 0, stream>>>(x, y, c1, c2,
                                                W1, b1, W2, b2, W3, b3, W4, b4,
                                                out, npts);
}
